// CP_Attention_87591563034972
// MI455X (gfx1250) — compile-verified, matrix operand order checked by reading, not run on hardware
//
#include <hip/hip_runtime.h>
#include <hip/hip_bf16.h>
#include <cstddef>
#include <cstdint>

// ---------------------------------------------------------------- constants
#define BSZ  8
#define SEQ  1024
#define CDIM 768
#define NH   12
#define HD   64
#define RNK  64
#define ROWS (BSZ * SEQ)     // 8192

typedef __bf16 v4bf  __attribute__((ext_vector_type(4)));
typedef __bf16 v8bf  __attribute__((ext_vector_type(8)));
typedef __bf16 v16bf __attribute__((ext_vector_type(16)));
typedef float  v8f   __attribute__((ext_vector_type(8)));

// -------------------------------------------------------------- helpers
// Load a 16-element bf16 WMMA fragment for this lane.
// Per ISA layout (16-bit A 16x32): elements 0..7 are 8 contiguous bf16 at p,
// elements 8..15 are 8 contiguous bf16 at p+16.  Caller supplies
// p = base + k0 + (lane>=16 ? 8 : 0).  Same pattern serves B when B is stored
// as Bt[n][k] (column-major B).
__device__ __forceinline__ v16bf load_frag(const __bf16* p) {
  v8bf lo = *reinterpret_cast<const v8bf*>(p);
  v8bf hi = *reinterpret_cast<const v8bf*>(p + 16);
  v16bf f;
#pragma unroll
  for (int i = 0; i < 8; ++i) { f[i] = lo[i]; f[8 + i] = hi[i]; }
  return f;
}

__device__ __forceinline__ v8f wmma_bf16(v16bf a, v16bf b, v8f c) {
  return __builtin_amdgcn_wmma_f32_16x16x32_bf16(
      false, a, false, b, (short)0, c, false, false);
}

__device__ __forceinline__ float hmax16(float v) {
#pragma unroll
  for (int m = 1; m < 16; m <<= 1) v = fmaxf(v, __shfl_xor(v, m, 32));
  return v;
}
__device__ __forceinline__ float hsum16(float v) {
#pragma unroll
  for (int m = 1; m < 16; m <<= 1) v += __shfl_xor(v, m, 32);
  return v;
}

// Async copy of 16 bytes global -> LDS (ASYNCcnt-tracked, gfx1250).
// lds_off: byte offset of destination within the workgroup LDS allocation
// (generic-pointer low 32 bits).  gaddr: 64-bit global source address.
__device__ __forceinline__ void async_ld16(uint32_t lds_off, const void* g) {
  asm volatile("global_load_async_to_lds_b128 %0, %1, off"
               :: "v"(lds_off), "v"((unsigned long long)(uintptr_t)g)
               : "memory");
}
__device__ __forceinline__ void wait_async0() {
  asm volatile("s_wait_asynccnt 0" ::: "memory");
}

// ---------------------------------------------------------------- kernel 1
// Build bf16 weights:
//  weff[s][r][c] = W1_s[c/HD, r] * W2_s[c%HD, r]        (Bt layout for stage-1)
//  w0bf[s][m][r] = W0_s[m][r]                            (native = Bt stage-2)
//  projbf[m][c]  = proj_w[m][c]                          (native = Bt final)
__global__ void k_prep(const float* wq0, const float* wq1, const float* wq2,
                       const float* wk0, const float* wk1, const float* wk2,
                       const float* wv0, const float* wv1, const float* wv2,
                       const float* projw,
                       __bf16* weff, __bf16* w0bf, __bf16* projbf) {
  const int S1 = 3 * RNK * CDIM;
  const int S2 = 3 * CDIM * RNK;
  const int S3 = CDIM * CDIM;
  int idx = blockIdx.x * blockDim.x + threadIdx.x;
  if (idx < S1) {
    int s = idx / (RNK * CDIM);
    int i = idx % (RNK * CDIM);
    int r = i / CDIM, c = i % CDIM;
    int n = c / HD, d = c % HD;
    const float* W1 = (s == 0) ? wq1 : (s == 1) ? wk1 : wv1;
    const float* W2 = (s == 0) ? wq2 : (s == 1) ? wk2 : wv2;
    weff[idx] = (__bf16)(W1[n * RNK + r] * W2[d * RNK + r]);
  } else if (idx < S1 + S2) {
    int j = idx - S1;
    int s = j / (CDIM * RNK);
    int i = j % (CDIM * RNK);
    const float* W0 = (s == 0) ? wq0 : (s == 1) ? wk0 : wv0;
    w0bf[j] = (__bf16)W0[i];
  } else if (idx < S1 + S2 + S3) {
    int i = idx - S1 - S2;
    projbf[i] = (__bf16)projw[i];
  }
}

// ---------------------------------------------------------------- kernel 2
__global__ void k_x2bf(const float* __restrict__ x, __bf16* __restrict__ xbf,
                       int n4) {
  int i = blockIdx.x * blockDim.x + threadIdx.x;
  if (i >= n4) return;
  float4 v = reinterpret_cast<const float4*>(x)[i];
  v4bf o;
  o[0] = (__bf16)v.x; o[1] = (__bf16)v.y; o[2] = (__bf16)v.z; o[3] = (__bf16)v.w;
  reinterpret_cast<v4bf*>(xbf)[i] = o;
}

// ---------------------------------------------------------------- kernel 3
// Fused low-rank projection: t = X @ Weff (K=768,N=64), out = t @ W0t (K=64,N=768)
// 128 rows / block, 8 waves, 16 rows / wave.  t staged through LDS in bf16.
// Stage-1 inner loop software-pipelined: next k-step's fragments are in
// flight while this k-step's 4 WMMAs execute.
__global__ __launch_bounds__(256) void k_lowrank(
    const __bf16* __restrict__ xbf, const __bf16* __restrict__ weff_all,
    const __bf16* __restrict__ w0_all, __bf16* __restrict__ qb,
    __bf16* __restrict__ kb, __bf16* __restrict__ vtb) {
  __shared__ __bf16 tbuf[128 * RNK];  // 16 KB
  const int sel = blockIdx.y;
  const __bf16* wf = weff_all + (size_t)sel * RNK * CDIM;
  const __bf16* w0 = w0_all + (size_t)sel * CDIM * RNK;
  const int w = threadIdx.x >> 5, lane = threadIdx.x & 31;
  const int li = lane & 15, hh = lane >> 4, base = hh * 8;
  const int row0 = blockIdx.x * 128 + w * 16;

  // ---- stage 1: t[16 x 64] = X[16 x 768] @ Weff (pipelined) ----
  v8f zero = {};
  v8f acc[4] = {zero, zero, zero, zero};
  const __bf16* arow = xbf + (size_t)(row0 + li) * CDIM;
  v16bf a = load_frag(arow + base);
  v16bf b0 = load_frag(wf + (size_t)(0 * 16 + li) * CDIM + base);
  v16bf b1 = load_frag(wf + (size_t)(1 * 16 + li) * CDIM + base);
  v16bf b2 = load_frag(wf + (size_t)(2 * 16 + li) * CDIM + base);
  v16bf b3 = load_frag(wf + (size_t)(3 * 16 + li) * CDIM + base);
  for (int k0 = 0; k0 < CDIM; k0 += 32) {
    const int k1 = (k0 + 32 < CDIM) ? (k0 + 32) : 0;  // wrap: harmless reload
    v16bf an = load_frag(arow + k1 + base);
    v16bf bn0 = load_frag(wf + (size_t)(0 * 16 + li) * CDIM + k1 + base);
    v16bf bn1 = load_frag(wf + (size_t)(1 * 16 + li) * CDIM + k1 + base);
    v16bf bn2 = load_frag(wf + (size_t)(2 * 16 + li) * CDIM + k1 + base);
    v16bf bn3 = load_frag(wf + (size_t)(3 * 16 + li) * CDIM + k1 + base);
    acc[0] = wmma_bf16(a, b0, acc[0]);
    acc[1] = wmma_bf16(a, b1, acc[1]);
    acc[2] = wmma_bf16(a, b2, acc[2]);
    acc[3] = wmma_bf16(a, b3, acc[3]);
    a = an; b0 = bn0; b1 = bn1; b2 = bn2; b3 = bn3;
  }
#pragma unroll
  for (int nt = 0; nt < 4; ++nt)
#pragma unroll
    for (int g = 0; g < 8; ++g)
      tbuf[(w * 16 + g + hh * 8) * RNK + nt * 16 + li] = (__bf16)acc[nt][g];
  __syncthreads();

  // ---- stage 2: out[16 x 768] = t @ W0t ----
  v16bf a0 = load_frag(&tbuf[(w * 16 + li) * RNK + base]);
  v16bf a1 = load_frag(&tbuf[(w * 16 + li) * RNK + 32 + base]);
  for (int ct = 0; ct < CDIM / 16; ++ct) {
    v8f c = zero;
    const int m = ct * 16 + li;
    v16bf bb0 = load_frag(w0 + (size_t)m * RNK + base);
    v16bf bb1 = load_frag(w0 + (size_t)m * RNK + 32 + base);
    c = wmma_bf16(a0, bb0, c);
    c = wmma_bf16(a1, bb1, c);
    const int h = m >> 6, d = m & 63;
#pragma unroll
    for (int g = 0; g < 8; ++g) {
      int grow = row0 + g + hh * 8;
      int bidx = grow >> 10, n = grow & 1023;
      __bf16 val = (__bf16)c[g];
      if (sel == 0)
        qb[(((size_t)bidx * NH + h) * SEQ + n) * HD + d] = val;
      else if (sel == 1)
        kb[(((size_t)bidx * NH + h) * SEQ + n) * HD + d] = val;
      else
        vtb[(((size_t)bidx * NH + h) * HD + d) * SEQ + n] = val;
    }
  }
}

// ---------------------------------------------------------------- kernel 4
// Flash attention, one (b,h,128-query-block) per CTA; 16 query rows per wave.
// K and V tiles (32 keys) are shared by all 8 waves: staged in LDS via
// double-buffered async global->LDS b128 copies (one per thread per tile),
// prefetching block j+1 while computing block j.
__global__ __launch_bounds__(256) void k_attn(const __bf16* __restrict__ qb,
                                              const __bf16* __restrict__ kb,
                                              const __bf16* __restrict__ vtb,
                                              __bf16* __restrict__ ybf) {
  __shared__ __bf16 ktile[2 * 32 * HD];   // 8 KB  [buf][key][d]
  __shared__ __bf16 vtile[2 * HD * 32];   // 8 KB  [buf][d][key]
  __shared__ __bf16 pbuf[8 * 16 * 32];    // 8 KB  per-wave P tiles
  const int bh = blockIdx.x >> 3;
  const int q0 = (blockIdx.x & 7) * 128;
  const int b = bh / NH, h = bh % NH;
  const __bf16* Q = qb + (size_t)bh * SEQ * HD;
  const __bf16* K = kb + (size_t)bh * SEQ * HD;
  const __bf16* Vt = vtb + (size_t)bh * HD * SEQ;
  const int tid = threadIdx.x;
  const int w = tid >> 5, lane = tid & 31;
  const int li = lane & 15, hh = lane >> 4, base = hh * 8;
  const float scale = 0.125f;  // 1/sqrt(64)

  // cooperative async tile fetch: K tile 32x64 (rows tid>>3), V tile 64x32
  auto prefetch = [&](int j0, int nb) {
    async_ld16(
        (uint32_t)(uintptr_t)&ktile[nb * (32 * HD) + (tid >> 3) * HD + (tid & 7) * 8],
        K + (size_t)(j0 + (tid >> 3)) * HD + (tid & 7) * 8);
    async_ld16(
        (uint32_t)(uintptr_t)&vtile[nb * (HD * 32) + (tid >> 2) * 32 + (tid & 3) * 8],
        Vt + (size_t)(tid >> 2) * SEQ + j0 + (tid & 3) * 8);
  };

  const int qrow = q0 + w * 16 + li;
  v16bf aq0 = load_frag(Q + (size_t)qrow * HD + base);
  v16bf aq1 = load_frag(Q + (size_t)qrow * HD + 32 + base);

  v8f zero = {};
  v8f o[4] = {zero, zero, zero, zero};
  float mrow[8], lrow[8];
#pragma unroll
  for (int g = 0; g < 8; ++g) { mrow[g] = -1e30f; lrow[g] = 0.f; }
  __bf16* pw = pbuf + w * 16 * 32;

  prefetch(0, 0);
  for (int it = 0; it < SEQ / 32; ++it) {
    const int bsel = it & 1;
    wait_async0();      // this wave's current-buffer copies have landed
    __syncthreads();    // all waves' copies landed; prev buffer fully consumed
    if (it + 1 < SEQ / 32) prefetch((it + 1) * 32, bsel ^ 1);

    const __bf16* kt = &ktile[bsel * (32 * HD)];
    const __bf16* vt = &vtile[bsel * (HD * 32)];

    // S tiles for the 32-key block (two 16x16 tiles), B frags from LDS
    v8f s0 = zero, s1 = zero;
    {
      v16bf bk0 = load_frag(kt + (size_t)li * HD + base);
      v16bf bk1 = load_frag(kt + (size_t)li * HD + 32 + base);
      s0 = wmma_bf16(aq0, bk0, s0);
      s0 = wmma_bf16(aq1, bk1, s0);
      v16bf bk2 = load_frag(kt + (size_t)(16 + li) * HD + base);
      v16bf bk3 = load_frag(kt + (size_t)(16 + li) * HD + 32 + base);
      s1 = wmma_bf16(aq0, bk2, s1);
      s1 = wmma_bf16(aq1, bk3, s1);
    }
    // online softmax update; each C-fragment row lives in one 16-lane half
#pragma unroll
    for (int g = 0; g < 8; ++g) {
      float x0 = s0[g] * scale, x1 = s1[g] * scale;
      float mx = hmax16(fmaxf(x0, x1));
      float mnew = fmaxf(mrow[g], mx);
      float alpha = __expf(mrow[g] - mnew);
      float p0 = __expf(x0 - mnew), p1 = __expf(x1 - mnew);
      float rs = hsum16(p0 + p1);
      lrow[g] = alpha * lrow[g] + rs;
      mrow[g] = mnew;
#pragma unroll
      for (int ct = 0; ct < 4; ++ct) o[ct][g] *= alpha;
      int r = g + hh * 8;
      pw[r * 32 + li] = (__bf16)p0;
      pw[r * 32 + 16 + li] = (__bf16)p1;
    }
    __syncthreads();  // uniform trip count; orders pbuf write -> A-frag read
    v16bf ap = load_frag(&pw[li * 32 + base]);
#pragma unroll
    for (int ct = 0; ct < 4; ++ct) {
      v16bf bv = load_frag(vt + (size_t)(ct * 16 + li) * 32 + base);
      o[ct] = wmma_bf16(ap, bv, o[ct]);
    }
  }

  // epilogue: normalize and write [B][N][CDIM] in bf16
#pragma unroll
  for (int ct = 0; ct < 4; ++ct)
#pragma unroll
    for (int g = 0; g < 8; ++g) {
      int n = q0 + w * 16 + g + hh * 8;
      int col = h * HD + ct * 16 + li;
      float val = o[ct][g] / lrow[g];
      ybf[((size_t)b * SEQ + n) * CDIM + col] = (__bf16)val;
    }
}

// ---------------------------------------------------------------- kernel 5
// out[8192 x 768] = ybf @ proj_w^T + proj_b  (projbf native [m][c] == Bt)
// Inner k-loop software-pipelined like k_lowrank stage 1.
__global__ __launch_bounds__(256) void k_final(const __bf16* __restrict__ ybf,
                                               const __bf16* __restrict__ projbf,
                                               const float* __restrict__ bias,
                                               float* __restrict__ out) {
  const int w = threadIdx.x >> 5, lane = threadIdx.x & 31;
  const int li = lane & 15, hh = lane >> 4, base = hh * 8;
  const int row0 = blockIdx.x * 128 + w * 16;
  const __bf16* arow = ybf + (size_t)(row0 + li) * CDIM;
  v8f zero = {};
  for (int ctg = 0; ctg < 12; ++ctg) {
    const int m0 = ctg * 64 + li;
    v8f acc[4] = {zero, zero, zero, zero};
    v16bf a = load_frag(arow + base);
    v16bf b0 = load_frag(projbf + (size_t)(m0 + 0) * CDIM + base);
    v16bf b1 = load_frag(projbf + (size_t)(m0 + 16) * CDIM + base);
    v16bf b2 = load_frag(projbf + (size_t)(m0 + 32) * CDIM + base);
    v16bf b3 = load_frag(projbf + (size_t)(m0 + 48) * CDIM + base);
    for (int k0 = 0; k0 < CDIM; k0 += 32) {
      const int k1 = (k0 + 32 < CDIM) ? (k0 + 32) : 0;  // wrap: harmless
      v16bf an = load_frag(arow + k1 + base);
      v16bf bn0 = load_frag(projbf + (size_t)(m0 + 0) * CDIM + k1 + base);
      v16bf bn1 = load_frag(projbf + (size_t)(m0 + 16) * CDIM + k1 + base);
      v16bf bn2 = load_frag(projbf + (size_t)(m0 + 32) * CDIM + k1 + base);
      v16bf bn3 = load_frag(projbf + (size_t)(m0 + 48) * CDIM + k1 + base);
      acc[0] = wmma_bf16(a, b0, acc[0]);
      acc[1] = wmma_bf16(a, b1, acc[1]);
      acc[2] = wmma_bf16(a, b2, acc[2]);
      acc[3] = wmma_bf16(a, b3, acc[3]);
      a = an; b0 = bn0; b1 = bn1; b2 = bn2; b3 = bn3;
    }
#pragma unroll
    for (int i = 0; i < 4; ++i) {
      int m = m0 + i * 16;
      float pb = bias[m];
#pragma unroll
      for (int g = 0; g < 8; ++g) {
        int grow = row0 + g + hh * 8;
        out[(size_t)grow * CDIM + m] = acc[i][g] + pb;
      }
    }
  }
}

// ---------------------------------------------------------------- launcher
extern "C" void kernel_launch(void* const* d_in, const int* in_sizes, int n_in,
                              void* d_out, int out_size, void* d_ws,
                              size_t ws_size, hipStream_t stream) {
  const float* x     = (const float*)d_in[0];
  const float* wq0   = (const float*)d_in[1];
  const float* wq1   = (const float*)d_in[2];
  const float* wq2   = (const float*)d_in[3];
  const float* wk0   = (const float*)d_in[4];
  const float* wk1   = (const float*)d_in[5];
  const float* wk2   = (const float*)d_in[6];
  const float* wv0   = (const float*)d_in[7];
  const float* wv1   = (const float*)d_in[8];
  const float* wv2   = (const float*)d_in[9];
  const float* projw = (const float*)d_in[10];
  const float* projb = (const float*)d_in[11];
  float* out = (float*)d_out;

  char* ws = (char*)d_ws;
  size_t off = 0;
  auto take = [&](size_t bytes) {
    size_t r = off;
    off = (off + bytes + 255) & ~(size_t)255;
    return r;
  };
  __bf16* xbf    = (__bf16*)(ws + take((size_t)ROWS * CDIM * 2));
  __bf16* weff   = (__bf16*)(ws + take((size_t)3 * RNK * CDIM * 2));
  __bf16* w0bf   = (__bf16*)(ws + take((size_t)3 * CDIM * RNK * 2));
  __bf16* projbf = (__bf16*)(ws + take((size_t)CDIM * CDIM * 2));
  __bf16* qb     = (__bf16*)(ws + take((size_t)BSZ * NH * SEQ * HD * 2));
  __bf16* kb     = (__bf16*)(ws + take((size_t)BSZ * NH * SEQ * HD * 2));
  __bf16* vtb    = (__bf16*)(ws + take((size_t)BSZ * NH * HD * SEQ * 2));
  __bf16* ybf    = (__bf16*)(ws + take((size_t)ROWS * CDIM * 2));
  (void)ws_size; (void)in_sizes; (void)n_in; (void)out_size;

  // 1) weight prep (3456 blocks covers 884736 elements exactly)
  k_prep<<<3456, 256, 0, stream>>>(wq0, wq1, wq2, wk0, wk1, wk2, wv0, wv1, wv2,
                                   projw, weff, w0bf, projbf);
  // 2) x -> bf16 (float4 at a time)
  k_x2bf<<<(ROWS * CDIM / 4 + 255) / 256, 256, 0, stream>>>(x, xbf,
                                                            ROWS * CDIM / 4);
  // 3) fused low-rank Q/K/V projection (grid.y selects Q/K/V)
  k_lowrank<<<dim3(ROWS / 128, 3), 256, 0, stream>>>(xbf, weff, w0bf, qb, kb,
                                                     vtb);
  // 4) flash attention with async double-buffered K/V tiles
  k_attn<<<BSZ * NH * (SEQ / 128), 256, 0, stream>>>(qb, kb, vtb, ybf);
  // 5) output projection + bias
  k_final<<<ROWS / 128, 256, 0, stream>>>(ybf, projbf, projb, out);
}
